// backbone_GIN_25752623907301
// MI455X (gfx1250) — matrix operand order repaired and checked by reading, not compiled, not run on hardware
//
#include <hip/hip_runtime.h>
#include <hip/hip_bf16.h>
#include <stdint.h>

// ---------------------------------------------------------------------------
// GIN backbone for MI455X (gfx1250, wave32).
// Dominant op: agg = adj(16384x16384, f32) @ h(16384x{128,64}) -> HBM bound on
// adj (1 GiB/layer).  v_wmma_f32_16x16x32_bf16 with:
//   - adj f32 -> packed bf16 in-register via v_perm_b32 (1 VALU/dword)
//   - B tile double-buffered in LDS; filled with GLOBAL_LOAD_ASYNC_TO_LDS_B128
//     (ASYNCcnt) when available, else cooperative b128 loads (no spills).
// ---------------------------------------------------------------------------

#if __has_builtin(__builtin_amdgcn_global_load_async_to_lds_b128)
#define GIN_ASYNC_LDS 1
#else
#define GIN_ASYNC_LDS 0
#endif

typedef __bf16 bf16x16 __attribute__((ext_vector_type(16)));
typedef float  v8f     __attribute__((ext_vector_type(8)));

// pointer types matching the async-LDS builtin's signature
// (param type per hipcc diagnostic: "__vector_size__(4*sizeof(int)) int __device__ *")
typedef int v4i_ __attribute__((vector_size(4 * sizeof(int))));
typedef __attribute__((address_space(1))) v4i_ gv4i;
typedef __attribute__((address_space(3))) v4i_ lv4i;

union FragU {
  unsigned u[8];
  uint4    q[2];
  bf16x16  v;
};

// pack two f32 into a bf16 pair (truncation) with a single v_perm_b32:
// result = {hi[31:16], lo[31:16]}
__device__ __forceinline__ unsigned pk_bf16(float lo, float hi) {
  return __builtin_amdgcn_perm(__builtin_bit_cast(unsigned, hi),
                               __builtin_bit_cast(unsigned, lo), 0x07060302u);
}
__device__ __forceinline__ unsigned short to_bf16(float x) {
  return (unsigned short)(__builtin_bit_cast(unsigned, x) >> 16);
}

__device__ __forceinline__ void wait_async0() {
#if __has_builtin(__builtin_amdgcn_s_wait_asynccnt)
  __builtin_amdgcn_s_wait_asynccnt(0);
#else
  asm volatile("s_wait_asynccnt 0x0" ::: "memory");
#endif
}

// A fragment (16x32 bf16) gathered from an f32 row-major matrix.
// `row` points at col 0 of this lane's M row (M = lane & 15).
// Layout per ISA 7.12.2: lanes<16 hold K 0..7 (v0-3) and 16..23 (v4-7);
// lanes>=16 hold K 8..15 and 24..31.
__device__ __forceinline__ void load_fragA_f32(int lane, const float* __restrict__ row,
                                               int k0, FragU& a) {
  const float* p = row + k0 + ((lane & 16) >> 1);   // +8 floats for upper half-wave
  float4 x0 = *(const float4*)(p + 0);
  float4 x1 = *(const float4*)(p + 4);
  float4 y0 = *(const float4*)(p + 16);
  float4 y1 = *(const float4*)(p + 20);
  a.u[0] = pk_bf16(x0.x, x0.y); a.u[1] = pk_bf16(x0.z, x0.w);
  a.u[2] = pk_bf16(x1.x, x1.y); a.u[3] = pk_bf16(x1.z, x1.w);
  a.u[4] = pk_bf16(y0.x, y0.y); a.u[5] = pk_bf16(y0.z, y0.w);
  a.u[6] = pk_bf16(y1.x, y1.y); a.u[7] = pk_bf16(y1.z, y1.w);
}

// B fragment (32x16 bf16) from an N-major (transposed) bf16 matrix in global:
// lanes<16 hold K 0..7 and 16..23, lanes>=16 hold K 8..15 and 24..31 (two 16B loads).
__device__ __forceinline__ void load_fragB_bf16(int lane, const unsigned short* __restrict__ row,
                                                int k0, FragU& b) {
  const unsigned short* p = row + k0 + ((lane & 16) >> 1);
  b.q[0] = *(const uint4*)(p + 0);
  b.q[1] = *(const uint4*)(p + 16);
}

// ---------------------------------------------------------------------------
// agg[N x DIN] = adj[N x N] @ h[N x DIN]   (h given feature-major bf16: hBT)
// 4 waves per WG, 16 output rows per wave, BM = 64 -> 256 workgroups.
// B tile (32 x DIN) staged in LDS with double buffering.
// ---------------------------------------------------------------------------
template<int DIN>
__global__ __launch_bounds__(128, 1) void gemm_agg_kernel(
    const float* __restrict__ adj, const unsigned short* __restrict__ hBT,
    float* __restrict__ agg) {
  constexpr int NT      = DIN / 16;
  constexpr int PER_THR = (DIN * 4) / 128;   // 16B chunks per thread per K-step
  __shared__ __align__(16) unsigned short ldsB[2][DIN * 32];

  const int tid  = threadIdx.x;
  const int lane = tid & 31;
  const int wave = tid >> 5;
  const int m0   = blockIdx.x * 64 + wave * 16;
  const float* arow = adj + (size_t)(m0 + (lane & 15)) * 16384;

  v8f zero = {0.f, 0.f, 0.f, 0.f, 0.f, 0.f, 0.f, 0.f};
  v8f acc[NT];
#pragma unroll
  for (int t = 0; t < NT; ++t) acc[t] = zero;

  const int col  = lane & 15;
  const int bofs = (lane & 16) >> 1;    // K half select within LDS row

#if GIN_ASYNC_LDS
  // ---- async DMA path: global -> LDS, no VGPR staging, tracked by ASYNCcnt
  auto fetchB = [&](int k0, int buf) {
#pragma unroll
    for (int j = 0; j < PER_THR; ++j) {
      int c = tid + j * 128;
      const unsigned short* g = hBT + (size_t)(c >> 2) * 16384 + k0 + (c & 3) * 8;
      unsigned short* l = &ldsB[buf][(c >> 2) * 32 + (c & 3) * 8];
      __builtin_amdgcn_global_load_async_to_lds_b128((gv4i*)g, (lv4i*)l, 0, 0);
    }
  };
  fetchB(0, 0);
  wait_async0();
  __syncthreads();
  for (int k0 = 0; k0 < 16384; k0 += 32) {
    const int buf = (k0 >> 5) & 1;
    fetchB((k0 + 32) & 16383, buf ^ 1);   // DMA next tile while we compute
    FragU a; load_fragA_f32(lane, arow, k0, a);
#pragma unroll
    for (int t = 0; t < NT; ++t) {
      FragU b;
      const unsigned short* p = &ldsB[buf][(t * 16 + col) * 32 + bofs];
      b.q[0] = *(const uint4*)(p + 0);
      b.q[1] = *(const uint4*)(p + 16);
      acc[t] = __builtin_amdgcn_wmma_f32_16x16x32_bf16(
          false, a.v, false, b.v, (short)0, acc[t], false, false);
    }
    wait_async0();                         // our fills of lds[buf^1] landed
    __syncthreads();                       // everyone's fills + reads done
  }
#else
  // ---- register-staged path (launch_bounds(,1) keeps pre[] out of scratch)
  uint4 pre[PER_THR];
  auto loadB = [&](int k0) {
#pragma unroll
    for (int j = 0; j < PER_THR; ++j) {
      int c = tid + j * 128;
      pre[j] = *(const uint4*)(hBT + (size_t)(c >> 2) * 16384 + k0 + (c & 3) * 8);
    }
  };
  auto storeB = [&](int buf) {
#pragma unroll
    for (int j = 0; j < PER_THR; ++j) {
      int c = tid + j * 128;
      *(uint4*)(&ldsB[buf][(c >> 2) * 32 + (c & 3) * 8]) = pre[j];
    }
  };
  loadB(0);
  storeB(0);
  for (int k0 = 0; k0 < 16384; k0 += 32) {
    const int buf = (k0 >> 5) & 1;
    loadB((k0 + 32) & 16383);     // prefetch next tile (wraps harmlessly at end)
    __syncthreads();              // lds[buf] ready for all waves
    FragU a; load_fragA_f32(lane, arow, k0, a);
#pragma unroll
    for (int t = 0; t < NT; ++t) {
      FragU b;
      const unsigned short* p = &ldsB[buf][(t * 16 + col) * 32 + bofs];
      b.q[0] = *(const uint4*)(p + 0);
      b.q[1] = *(const uint4*)(p + 16);
      acc[t] = __builtin_amdgcn_wmma_f32_16x16x32_bf16(
          false, a.v, false, b.v, (short)0, acc[t], false, false);
    }
    storeB(buf ^ 1);              // fill the other buffer for next iteration
  }
#endif

  const int rofs = (lane & 16) >> 1;   // D-matrix: M = r + 8 for upper half-wave
#pragma unroll
  for (int t = 0; t < NT; ++t)
#pragma unroll
    for (int r = 0; r < 8; ++r)
      agg[(size_t)(m0 + rofs + r) * DIN + t * 16 + col] = acc[t][r];
}

// ---------------------------------------------------------------------------
// z = leaky_relu(agg @ W1 + b1) @ W2 + b2 ; accumulate per-feature sum/sumsq.
// Intermediate C-layout -> A-layout reshuffle goes through LDS.
// ---------------------------------------------------------------------------
template<int DIN>
__global__ __launch_bounds__(128) void mlp_kernel(
    const float* __restrict__ agg,
    const unsigned short* __restrict__ W1T, const float* __restrict__ b1,
    const unsigned short* __restrict__ W2T, const float* __restrict__ b2,
    float* __restrict__ z, float* __restrict__ S1g, float* __restrict__ S2g) {
  __shared__ __align__(16) unsigned short tbuf[4][16][64];
  __shared__ float S1loc[64];
  __shared__ float S2loc[64];

  const int tid = threadIdx.x;
  if (tid < 64) { S1loc[tid] = 0.f; S2loc[tid] = 0.f; }

  const int lane = tid & 31;
  const int wave = tid >> 5;
  const int m0   = blockIdx.x * 64 + wave * 16;
  const int col  = lane & 15;
  const int rofs = (lane & 16) >> 1;
  const float* arow = agg + (size_t)(m0 + col) * DIN;

  v8f zero = {0.f, 0.f, 0.f, 0.f, 0.f, 0.f, 0.f, 0.f};
  v8f acc[4];
#pragma unroll
  for (int t = 0; t < 4; ++t) acc[t] = zero;

#pragma unroll
  for (int k0 = 0; k0 < DIN; k0 += 32) {
    FragU a; load_fragA_f32(lane, arow, k0, a);
#pragma unroll
    for (int t = 0; t < 4; ++t) {
      FragU b; load_fragB_bf16(lane, W1T + (size_t)(t * 16 + col) * DIN, k0, b);
      acc[t] = __builtin_amdgcn_wmma_f32_16x16x32_bf16(
          false, a.v, false, b.v, (short)0, acc[t], false, false);
    }
  }

  __syncthreads();   // S1loc/S2loc init visible; about to fill tbuf

#pragma unroll
  for (int t = 0; t < 4; ++t) {
    float bv = b1[t * 16 + col];
#pragma unroll
    for (int r = 0; r < 8; ++r) {
      float v = acc[t][r] + bv;
      v = v > 0.f ? v : 0.01f * v;                 // leaky_relu(0.01)
      tbuf[wave][rofs + r][t * 16 + col] = to_bf16(v);
    }
  }
  __syncthreads();

  v8f acc2[4];
#pragma unroll
  for (int t = 0; t < 4; ++t) acc2[t] = zero;

#pragma unroll
  for (int k0 = 0; k0 < 64; k0 += 32) {
    FragU a;
    const unsigned short* p = &tbuf[wave][col][k0 + ((lane & 16) >> 1)];
    a.q[0] = *(const uint4*)(p + 0);
    a.q[1] = *(const uint4*)(p + 16);
#pragma unroll
    for (int t = 0; t < 4; ++t) {
      FragU b; load_fragB_bf16(lane, W2T + (size_t)(t * 16 + col) * 64, k0, b);
      acc2[t] = __builtin_amdgcn_wmma_f32_16x16x32_bf16(
          false, a.v, false, b.v, (short)0, acc2[t], false, false);
    }
  }

#pragma unroll
  for (int t = 0; t < 4; ++t) {
    float bv = b2[t * 16 + col];
    float s1 = 0.f, s2 = 0.f;
#pragma unroll
    for (int r = 0; r < 8; ++r) {
      float v = acc2[t][r] + bv;
      z[(size_t)(m0 + rofs + r) * 64 + t * 16 + col] = v;
      s1 += v; s2 += v * v;
    }
    atomicAdd(&S1loc[t * 16 + col], s1);
    atomicAdd(&S2loc[t * 16 + col], s2);
  }
  __syncthreads();
  if (tid < 64) {
    atomicAdd(&S1g[tid], S1loc[tid]);
    atomicAdd(&S2g[tid], S2loc[tid]);
  }
}

// ---------------------------------------------------------------------------
// BatchNorm (affine=False, biased var) + l_enc write + segment-sum pooling
// into g_enc (f32 global atomics) + next layer's feature-major bf16 h.
// ---------------------------------------------------------------------------
__global__ __launch_bounds__(256) void bnpool_kernel(
    const float* __restrict__ z, const float* __restrict__ S1g,
    const float* __restrict__ S2g, const int* __restrict__ batch,
    float* __restrict__ dout, unsigned short* __restrict__ hBT, int layer) {
  int i = blockIdx.x * 256 + threadIdx.x;     // exactly 16384*64 threads
  int n = i >> 6;
  int f = i & 63;
  const float invN = 1.0f / 16384.0f;
  float mean = S1g[f] * invN;
  float var  = S2g[f] * invN - mean * mean;
  float y = (z[i] - mean) * rsqrtf(var + 1e-4f);
  // l_enc [N,192] after g_enc [512,192]
  dout[(size_t)512 * 192 + (size_t)n * 192 + layer * 64 + f] = y;
  atomicAdd(&dout[(size_t)batch[n] * 192 + layer * 64 + f], y);
  hBT[(size_t)f * 16384 + n] = to_bf16(y);
}

__global__ __launch_bounds__(256) void prep_x_kernel(
    const float* __restrict__ x, unsigned short* __restrict__ hBT) {
  int i = blockIdx.x * 256 + threadIdx.x;     // 16384*128 threads
  int n = i >> 7;
  int f = i & 127;
  hBT[(size_t)f * 16384 + n] = to_bf16(x[i]);
}

__global__ __launch_bounds__(256) void prep_w_kernel(
    const float* __restrict__ W1a, const float* __restrict__ W1b,
    const float* __restrict__ W1c, const float* __restrict__ W2a,
    const float* __restrict__ W2b, const float* __restrict__ W2c,
    unsigned short* __restrict__ wbuf) {
  int i = blockIdx.x * 256 + threadIdx.x;     // 28672 threads
  if (i < 8192)       { int o = i / 128,          k = i % 128;          wbuf[i] = to_bf16(W1a[k * 64 + o]); }
  else if (i < 12288) { int j = i - 8192;  int o = j / 64, k = j % 64;  wbuf[i] = to_bf16(W1b[k * 64 + o]); }
  else if (i < 16384) { int j = i - 12288; int o = j / 64, k = j % 64;  wbuf[i] = to_bf16(W1c[k * 64 + o]); }
  else if (i < 20480) { int j = i - 16384; int o = j / 64, k = j % 64;  wbuf[i] = to_bf16(W2a[k * 64 + o]); }
  else if (i < 24576) { int j = i - 20480; int o = j / 64, k = j % 64;  wbuf[i] = to_bf16(W2b[k * 64 + o]); }
  else if (i < 28672) { int j = i - 24576; int o = j / 64, k = j % 64;  wbuf[i] = to_bf16(W2c[k * 64 + o]); }
}

__global__ __launch_bounds__(256) void init_kernel(float* __restrict__ dout,
                                                   float* __restrict__ S) {
  int i = blockIdx.x * 256 + threadIdx.x;
  if (i < 512 * 192) dout[i] = 0.f;           // zero g_enc (atomics accumulate)
  else if (i < 512 * 192 + 384) S[i - 512 * 192] = 0.f;  // zero BN accumulators
}

// ---------------------------------------------------------------------------
extern "C" void kernel_launch(void* const* d_in, const int* in_sizes, int n_in,
                              void* d_out, int out_size, void* d_ws, size_t ws_size,
                              hipStream_t stream) {
  const float* x   = (const float*)d_in[0];
  const float* adj = (const float*)d_in[1];
  const float* W1[3] = {(const float*)d_in[2], (const float*)d_in[6],  (const float*)d_in[10]};
  const float* b1[3] = {(const float*)d_in[3], (const float*)d_in[7],  (const float*)d_in[11]};
  const float* W2[3] = {(const float*)d_in[4], (const float*)d_in[8],  (const float*)d_in[12]};
  const float* b2[3] = {(const float*)d_in[5], (const float*)d_in[9],  (const float*)d_in[13]};
  const int*   batch = (const int*)d_in[14];
  float* dout = (float*)d_out;

  char* ws = (char*)d_ws;
  unsigned short* hBT  = (unsigned short*)(ws);                       // 128*16384*2 = 4 MiB
  float*          agg  = (float*)(ws + (size_t)(4u  << 20));          // 16384*128*4 = 8 MiB
  float*          zbuf = (float*)(ws + (size_t)(12u << 20));          // 16384*64*4  = 4 MiB
  float*          S    = (float*)(ws + (size_t)(16u << 20));          // 6*64 floats
  unsigned short* wbuf = (unsigned short*)(ws + (size_t)(16u << 20) + 4096); // 56 KiB

  unsigned short* W1T[3] = {wbuf, wbuf + 8192, wbuf + 12288};
  unsigned short* W2T[3] = {wbuf + 16384, wbuf + 20480, wbuf + 24576};

  init_kernel  <<<386,  256, 0, stream>>>(dout, S);
  prep_x_kernel<<<8192, 256, 0, stream>>>(x, hBT);
  prep_w_kernel<<<112,  256, 0, stream>>>(W1[0], W1[1], W1[2], W2[0], W2[1], W2[2], wbuf);

  for (int l = 0; l < 3; ++l) {
    float* S1 = S + l * 128;
    float* S2 = S + l * 128 + 64;
    if (l == 0) {
      gemm_agg_kernel<128><<<256, 128, 0, stream>>>(adj, hBT, agg);
      mlp_kernel<128><<<256, 128, 0, stream>>>(agg, W1T[0], b1[0], W2T[0], b2[0], zbuf, S1, S2);
    } else {
      gemm_agg_kernel<64><<<256, 128, 0, stream>>>(adj, hBT, agg);
      mlp_kernel<64><<<256, 128, 0, stream>>>(agg, W1T[l], b1[l], W2T[l], b2[l], zbuf, S1, S2);
    }
    bnpool_kernel<<<4096, 256, 0, stream>>>(zbuf, S1, S2, batch, dout, hBT, l);
  }
  (void)in_sizes; (void)n_in; (void)out_size; (void)ws_size;
}
